// LSTMModel_77670188581192
// MI455X (gfx1250) — matrix operand order repaired and checked by reading, not compiled, not run on hardware
//
#include <hip/hip_runtime.h>
#include <hip/hip_bf16.h>

#define B_SZ 256
#define T_SZ 128
#define D_SZ 32
#define H_SZ 512
#define L_SZ 6
#define G4   (4 * H_SZ)   // 2048
#define EPSV 1e-5f

typedef __attribute__((ext_vector_type(8)))  float  v8f;
typedef __attribute__((ext_vector_type(8)))  __bf16 v8bf;
typedef __attribute__((ext_vector_type(16))) __bf16 v16bf;

// ---------- helpers ----------------------------------------------------------

__device__ inline unsigned short f32_to_bf16u(float f) {
  union { float f; unsigned u; } v; v.f = f;
  unsigned u = v.u;
  unsigned r = u + 0x7FFFu + ((u >> 16) & 1u);  // round-to-nearest-even
  return (unsigned short)(r >> 16);
}

__device__ inline float fast_sigmoid(float x) {
  return 1.0f / (1.0f + __expf(-x));
}
__device__ inline float fast_tanh(float x) {
  x = fminf(fmaxf(x, -15.0f), 15.0f);
  float e = __expf(2.0f * x);
  return (e - 1.0f) / (e + 1.0f);
}

// A-fragment (16x32, 16-bit): lane half 0 holds K {k0..k0+15},
// half 1 holds K {k0+16..k0+31}; row = m0 + (lane&15). Two 16B loads.
__device__ inline v16bf load_a_frag(const unsigned short* A, int lda, int m0, int k0) {
  int lane = threadIdx.x & 31;
  int half = lane >> 4;
  const unsigned short* base = A + (size_t)(m0 + (lane & 15)) * lda + k0 + half * 16;
  v8bf lo = *(const v8bf*)base;
  v8bf hi = *(const v8bf*)(base + 8);
  v16bf r;
#pragma unroll
  for (int i = 0; i < 8; ++i) { r[i] = lo[i]; r[8 + i] = hi[i]; }
  return r;
}

// B-fragment for C = A @ W^T, W row-major (N x K): B[k][n] = W[n][k].
// Lane n = n0+(lane&15); half selects K {k0..k0+15} / {k0+16..k0+31}:
// 16 contiguous bf16 along W's K axis -> two 16B loads.
__device__ inline v16bf load_bT_frag(const unsigned short* W, int ldw, int n0, int k0) {
  int lane = threadIdx.x & 31;
  int half = lane >> 4;
  const unsigned short* base = W + (size_t)(n0 + (lane & 15)) * ldw + k0 + half * 16;
  v8bf lo = *(const v8bf*)base;
  v8bf hi = *(const v8bf*)(base + 8);
  v16bf r;
#pragma unroll
  for (int i = 0; i < 8; ++i) { r[i] = lo[i]; r[8 + i] = hi[i]; }
  return r;
}

// Device-wide sense-reversal barrier (persistent-kernel scan).
__device__ inline void grid_barrier(unsigned* cnt, unsigned* gen, unsigned nblocks) {
  __syncthreads();
  if (threadIdx.x == 0) {
    __builtin_amdgcn_fence(__ATOMIC_RELEASE, "agent");
    unsigned g = __hip_atomic_load(gen, __ATOMIC_RELAXED, __HIP_MEMORY_SCOPE_AGENT);
    unsigned arrived = __hip_atomic_fetch_add(cnt, 1u, __ATOMIC_ACQ_REL, __HIP_MEMORY_SCOPE_AGENT);
    if (arrived == nblocks - 1u) {
      __hip_atomic_store(cnt, 0u, __ATOMIC_RELAXED, __HIP_MEMORY_SCOPE_AGENT);
      __hip_atomic_store(gen, g + 1u, __ATOMIC_RELEASE, __HIP_MEMORY_SCOPE_AGENT);
    } else {
      while (__hip_atomic_load(gen, __ATOMIC_RELAXED, __HIP_MEMORY_SCOPE_AGENT) == g) {
        __builtin_amdgcn_s_sleep(2);
      }
    }
  }
  __syncthreads();
  __builtin_amdgcn_fence(__ATOMIC_ACQUIRE, "agent");
}

// ---------- kernels ----------------------------------------------------------

__global__ void init_bar_kernel(unsigned* p) {
  if (threadIdx.x < 2) p[threadIdx.x] = 0u;
}

__global__ __launch_bounds__(256) void cvt_bf16_kernel(
    const float* __restrict__ a, const float* __restrict__ b,
    unsigned short* __restrict__ oa, unsigned short* __restrict__ ob, int n) {
  int i = blockIdx.x * blockDim.x + threadIdx.x;
  int stride = gridDim.x * blockDim.x;
  for (; i < n; i += stride) {
    oa[i] = f32_to_bf16u(a[i]);
    ob[i] = f32_to_bf16u(b[i]);
  }
}

// h_prev = x @ w_proj^T + b_proj   (K = 32, memory bound -> scalar VALU)
__global__ __launch_bounds__(256) void proj_kernel(
    const float* __restrict__ x, const float* __restrict__ wp, const float* __restrict__ bp,
    float* __restrict__ hp, unsigned short* __restrict__ hp_bf) {
  size_t idx = (size_t)blockIdx.x * blockDim.x + threadIdx.x;
  size_t total = (size_t)B_SZ * T_SZ * H_SZ;
  if (idx >= total) return;
  int col = (int)(idx % H_SZ);
  size_t bt = idx / H_SZ;
  const float* xr = x + bt * D_SZ;
  const float* wr = wp + (size_t)col * D_SZ;
  float s = bp[col];
#pragma unroll
  for (int d = 0; d < D_SZ; ++d) s = fmaf(xr[d], wr[d], s);
  hp[idx] = s;
  hp_bf[idx] = f32_to_bf16u(s);
}

// xg = h_prev @ w_ih^T + (b_ih + b_hh) : M=32768 N=2048 K=512, bf16 WMMA.
// Each wave: 64x16 tile (4 M-tiles share one B fragment per K-step).
__global__ __launch_bounds__(256) void xg_gemm_kernel(
    const unsigned short* __restrict__ A,   // 32768 x 512 bf16
    const unsigned short* __restrict__ W,   // 2048 x 512 bf16 (layer slice)
    const float* __restrict__ bih, const float* __restrict__ bhh,
    float* __restrict__ xg) {               // 32768 x 2048 f32
  int wave = blockIdx.x * (blockDim.x >> 5) + (threadIdx.x >> 5);
  int n_tile = wave & 127;       // 128 N-tiles
  int m_super = wave >> 7;       // 512 M-supertiles of 64 rows
  int n0 = n_tile * 16;
  int m_base = m_super * 64;
  int lane = threadIdx.x & 31;
  int col = n0 + (lane & 15);
  float bias = bih[col] + bhh[col];

  v8f acc[4];
#pragma unroll
  for (int s = 0; s < 4; ++s)
#pragma unroll
    for (int r = 0; r < 8; ++r) acc[s][r] = bias;

  for (int k0 = 0; k0 < H_SZ; k0 += 32) {
    v16bf bf = load_bT_frag(W, H_SZ, n0, k0);
#pragma unroll
    for (int s = 0; s < 4; ++s) {
      v16bf af = load_a_frag(A, H_SZ, m_base + s * 16, k0);
      acc[s] = __builtin_amdgcn_wmma_f32_16x16x32_bf16(
          false, af, false, bf, (short)0, acc[s], false, false);
    }
  }
  int row_off = (lane >> 4) * 8;
#pragma unroll
  for (int s = 0; s < 4; ++s)
#pragma unroll
    for (int r = 0; r < 8; ++r)
      xg[(size_t)(m_base + s * 16 + r + row_off) * G4 + col] = acc[s][r];
}

// Persistent recurrence: 64 blocks x 8 waves = 512 waves; wave owns
// (16 batch x 16 hidden) tile and all 4 gate tiles; c-state in registers.
// w_hh fragments staged in LDS (64KB) in WMMA fragment order, once per layer.
__global__ __launch_bounds__(256) void lstm_scan_kernel(
    const float* __restrict__ xg,            // 32768 x 2048 f32
    const unsigned short* __restrict__ whh,  // 2048 x 512 bf16 (layer slice)
    unsigned short* __restrict__ hbuf0,      // 256 x 512 bf16 (ping)
    unsigned short* __restrict__ hbuf1,      // 256 x 512 bf16 (pong)
    float* __restrict__ out_pre,             // 256 x 128 x 512 f32 (pre-LN h_t)
    unsigned* bar_cnt, unsigned* bar_gen) {
  __shared__ unsigned short ldsB[4 * 16 * 32 * 16];  // [gate][kstep][lane][16] = 64KB

  int wid  = threadIdx.x >> 5;
  int lane = threadIdx.x & 31;
  int half = lane >> 4;
  int n_tile = blockIdx.x & 31;            // block shares hidden-column tile
  int m_tile = (blockIdx.x >> 5) * 8 + wid;
  int n0 = n_tile * 16;
  int m0 = m_tile * 16;
  int row_off = half * 8;
  int colh = n0 + (lane & 15);

  // Stage B fragments: wave w covers gate (w&3), ksteps [(w>>2)*8, +8)
  {
    int g = wid & 3;
    int ks0 = (wid >> 2) * 8;
    for (int ks = ks0; ks < ks0 + 8; ++ks) {
      const unsigned short* src =
          whh + (size_t)(g * H_SZ + n0 + (lane & 15)) * H_SZ + ks * 32 + half * 16;
      v8bf lo = *(const v8bf*)src;
      v8bf hi = *(const v8bf*)(src + 8);
      unsigned short* dst = &ldsB[(((g * 16 + ks) * 32) + lane) * 16];
      *(v8bf*)dst = lo;
      *(v8bf*)(dst + 8) = hi;
    }
  }
  __syncthreads();

  // h_0 = 0
#pragma unroll
  for (int r = 0; r < 8; ++r)
    hbuf0[(size_t)(m0 + r + row_off) * H_SZ + colh] = 0;
  grid_barrier(bar_cnt, bar_gen, gridDim.x);

  v8f c_state;
#pragma unroll
  for (int r = 0; r < 8; ++r) c_state[r] = 0.0f;

  for (int t = 0; t < T_SZ; ++t) {
    const unsigned short* hcur = (t & 1) ? hbuf1 : hbuf0;
    unsigned short*       hnxt = (t & 1) ? hbuf0 : hbuf1;

    // init gates from precomputed xg (includes both biases)
    v8f acc[4];
#pragma unroll
    for (int g = 0; g < 4; ++g)
#pragma unroll
      for (int r = 0; r < 8; ++r)
        acc[g][r] = xg[((size_t)(m0 + r + row_off) * T_SZ + t) * G4 + g * H_SZ + colh];

    // gates += h_{t-1} @ w_hh^T
    for (int ks = 0; ks < 16; ++ks) {
      v16bf af = load_a_frag(hcur, H_SZ, m0, ks * 32);
#pragma unroll
      for (int g = 0; g < 4; ++g) {
        const unsigned short* bp = &ldsB[(((g * 16 + ks) * 32) + lane) * 16];
        v8bf lo = *(const v8bf*)bp;
        v8bf hi = *(const v8bf*)(bp + 8);
        v16bf bf;
#pragma unroll
        for (int i = 0; i < 8; ++i) { bf[i] = lo[i]; bf[8 + i] = hi[i]; }
        acc[g] = __builtin_amdgcn_wmma_f32_16x16x32_bf16(
            false, af, false, bf, (short)0, acc[g], false, false);
      }
    }

    // lane-local LSTM cell update (i,f,g,o share (row,col) layout)
#pragma unroll
    for (int r = 0; r < 8; ++r) {
      float iv = fast_sigmoid(acc[0][r]);
      float fv = fast_sigmoid(acc[1][r]);
      float gv = fast_tanh(acc[2][r]);
      float ov = fast_sigmoid(acc[3][r]);
      float c = fmaf(fv, c_state[r], iv * gv);
      c_state[r] = c;
      float h = ov * fast_tanh(c);
      int brow = m0 + r + row_off;
      out_pre[((size_t)brow * T_SZ + t) * H_SZ + colh] = h;
      hnxt[(size_t)brow * H_SZ + colh] = f32_to_bf16u(h);
    }
    grid_barrier(bar_cnt, bar_gen, gridDim.x);
  }
}

// LayerNorm + residual; wave per row, shuffle reductions (wave32).
__global__ __launch_bounds__(256) void ln_kernel(
    const float* __restrict__ xin, const float* __restrict__ g, const float* __restrict__ bta,
    float* __restrict__ hp, unsigned short* __restrict__ hp_bf, int layer) {
  int wave = blockIdx.x * (blockDim.x >> 5) + (threadIdx.x >> 5);
  int lane = threadIdx.x & 31;
  size_t row = (size_t)wave;                 // 32768 rows total
  const float* xr = xin + row * H_SZ;
  float v[16];
  float s = 0.0f;
#pragma unroll
  for (int i = 0; i < 16; ++i) { v[i] = xr[lane + i * 32]; s += v[i]; }
#pragma unroll
  for (int m = 16; m >= 1; m >>= 1) s += __shfl_xor(s, m, 32);
  float mu = s * (1.0f / H_SZ);
  float q = 0.0f;
#pragma unroll
  for (int i = 0; i < 16; ++i) { float d = v[i] - mu; q = fmaf(d, d, q); }
#pragma unroll
  for (int m = 16; m >= 1; m >>= 1) q += __shfl_xor(q, m, 32);
  float rstd = rsqrtf(q * (1.0f / H_SZ) + EPSV);
#pragma unroll
  for (int i = 0; i < 16; ++i) {
    int c = lane + i * 32;
    float y = (v[i] - mu) * rstd * g[c] + bta[c];
    size_t idx = row * H_SZ + c;
    if (layer > 0) y += hp[idx];
    hp[idx] = y;
    hp_bf[idx] = f32_to_bf16u(y);
  }
}

// Head MLP: relu(512->512) -> relu(512->256) -> 256->1. One block per batch row.
__global__ __launch_bounds__(256) void head_kernel(
    const float* __restrict__ hp,
    const float* __restrict__ w1, const float* __restrict__ b1,
    const float* __restrict__ w2, const float* __restrict__ b2,
    const float* __restrict__ w3, const float* __restrict__ b3,
    float* __restrict__ out) {
  __shared__ float sl[H_SZ];
  __shared__ float z1[H_SZ];
  __shared__ float red[256];
  int b = blockIdx.x, tid = threadIdx.x;
  const float* row = hp + ((size_t)b * T_SZ + (T_SZ - 1)) * H_SZ;
  sl[tid] = row[tid];
  sl[tid + 256] = row[tid + 256];
  __syncthreads();
  for (int o = tid; o < H_SZ; o += 256) {
    float s = b1[o];
    const float* wr = w1 + (size_t)o * H_SZ;
    for (int k = 0; k < H_SZ; ++k) s = fmaf(wr[k], sl[k], s);
    z1[o] = fmaxf(s, 0.0f);
  }
  __syncthreads();
  float z2v;
  {
    float s = b2[tid];
    const float* wr = w2 + (size_t)tid * H_SZ;
    for (int k = 0; k < H_SZ; ++k) s = fmaf(wr[k], z1[k], s);
    z2v = fmaxf(s, 0.0f);
  }
  red[tid] = z2v * w3[tid];
  __syncthreads();
  for (int s = 128; s > 0; s >>= 1) {
    if (tid < s) red[tid] += red[tid + s];
    __syncthreads();
  }
  if (tid == 0) out[b] = red[0] + b3[0];
}

// ---------- host launcher ----------------------------------------------------

extern "C" void kernel_launch(void* const* d_in, const int* in_sizes, int n_in,
                              void* d_out, int out_size, void* d_ws, size_t ws_size,
                              hipStream_t stream) {
  (void)in_sizes; (void)n_in; (void)out_size; (void)ws_size;
  const float* x      = (const float*)d_in[0];
  const float* w_proj = (const float*)d_in[1];
  const float* b_proj = (const float*)d_in[2];
  const float* w_ih   = (const float*)d_in[3];
  const float* w_hh   = (const float*)d_in[4];
  const float* b_ih   = (const float*)d_in[5];
  const float* b_hh   = (const float*)d_in[6];
  const float* ln_g   = (const float*)d_in[7];
  const float* ln_b   = (const float*)d_in[8];
  const float* w1     = (const float*)d_in[9];
  const float* b1     = (const float*)d_in[10];
  const float* w2     = (const float*)d_in[11];
  const float* b2     = (const float*)d_in[12];
  const float* w3     = (const float*)d_in[13];
  const float* b3     = (const float*)d_in[14];

  char* ws = (char*)d_ws;
  size_t off = 0;
  auto take = [&](size_t bytes) -> char* {
    char* p = ws + off;
    off += (bytes + 255) & ~(size_t)255;
    return p;
  };
  const size_t wcnt = (size_t)L_SZ * G4 * H_SZ;          // 6,291,456
  const size_t acnt = (size_t)B_SZ * T_SZ * H_SZ;        // 16,777,216
  unsigned short* wih_bf = (unsigned short*)take(wcnt * 2);
  unsigned short* whh_bf = (unsigned short*)take(wcnt * 2);
  float*          hp     = (float*)take(acnt * 4);       // h_prev fp32
  unsigned short* hp_bf  = (unsigned short*)take(acnt * 2);
  float*          outpre = (float*)take(acnt * 4);       // pre-LN scan output
  float*          xg     = (float*)take((size_t)B_SZ * T_SZ * G4 * 4);
  unsigned short* hbuf0  = (unsigned short*)take((size_t)B_SZ * H_SZ * 2);
  unsigned short* hbuf1  = (unsigned short*)take((size_t)B_SZ * H_SZ * 2);
  unsigned*       bar    = (unsigned*)take(256);

  init_bar_kernel<<<1, 64, 0, stream>>>(bar);
  cvt_bf16_kernel<<<(int)((wcnt + 255) / 256), 256, 0, stream>>>(
      w_ih, w_hh, wih_bf, whh_bf, (int)wcnt);
  proj_kernel<<<(int)((acnt + 255) / 256), 256, 0, stream>>>(x, w_proj, b_proj, hp, hp_bf);

  for (int l = 0; l < L_SZ; ++l) {
    const unsigned short* wih_l = wih_bf + (size_t)l * G4 * H_SZ;
    const unsigned short* whh_l = whh_bf + (size_t)l * G4 * H_SZ;
    // 65536 waves / 8 per block = 8192 blocks
    xg_gemm_kernel<<<8192, 256, 0, stream>>>(hp_bf, wih_l, b_ih + l * G4, b_hh + l * G4, xg);
    // persistent scan: 64 blocks (all resident), 1 grid barrier per timestep
    lstm_scan_kernel<<<64, 256, 0, stream>>>(xg, whh_l, hbuf0, hbuf1, outpre, bar, bar + 1);
    // 32768 rows / 8 waves per block = 4096 blocks
    ln_kernel<<<4096, 256, 0, stream>>>(outpre, ln_g + l * H_SZ, ln_b + l * H_SZ, hp, hp_bf, l);
  }
  head_kernel<<<B_SZ, 256, 0, stream>>>(hp, w1, b1, w2, b2, w3, b3, (float*)d_out);
}
